// ConditionalNSF_67834713473196
// MI455X (gfx1250) — matrix operand order repaired and checked by reading, not compiled, not run on hardware
//
#include <hip/hip_runtime.h>
#include <hip/hip_bf16.h>
#include <math.h>

typedef _Float16 f16x16 __attribute__((ext_vector_type(16)));
typedef float    f32x8  __attribute__((ext_vector_type(8)));

#define B_    16384
#define SD_   128
#define AD_   32
#define HID_  512
#define CTX_  512
#define H_    256
#define FD_   8
#define MP_   23      /* 3*K-1 */
#define NB_   2
#define L_    4
#define NPAD_ 192     /* FD_*MP_=184 padded to multiple of 64 */
#define TBND  6.0f
#define LNEPS 1e-5f

static __device__ __forceinline__ float softplusf(float x){
  return (x > 20.f) ? x : log1pf(expf(x));
}

// Async global->LDS copy (CDNA5 Tensor/Async path, ASYNCcnt-tracked).
// vdst = per-lane LDS byte offset, vaddr = 64-bit global address (GV mode).
static __device__ __forceinline__ void async_ld_b128(unsigned lds_off, const _Float16* g){
  asm volatile("global_load_async_to_lds_b128 %0, %1, off"
               :: "v"(lds_off), "v"(g) : "memory");
}

// ---------------------------------------------------------------------------
// WMMA GEMM: C[M x N] (f32) = A[M x K] (f16, row major) @ W[N x K]^T (f16)
// block tile 128(M) x 64(N), K step 32. 256 threads = 8 waves (wave32).
// Double-buffered LDS tiles filled by async global->LDS loads so the next
// tile streams in (ASYNCcnt) underneath the 4 v_wmma_f32_16x16x32_f16.
// ---------------------------------------------------------------------------
#define LDA_PAD 40   // halves per LDS row (32 + 8 skew)

__global__ __launch_bounds__(256) void k_gemm(
    const _Float16* __restrict__ A, const _Float16* __restrict__ W,
    float* __restrict__ C, int M, int N, int K)
{
  __shared__ __align__(16) _Float16 As[2][128 * LDA_PAD];
  __shared__ __align__(16) _Float16 Bs[2][64 * LDA_PAD];

  const int tid   = threadIdx.x;
  const int lane  = tid & 31;
  const int wave  = tid >> 5;
  const int l16   = lane & 15;
  const int lhalf = (lane >> 4) & 1;   // 0: lanes 0-15, 1: lanes 16-31
  const int m0 = blockIdx.y * 128;
  const int n0 = blockIdx.x * 64;
  const int wm = (wave & 3) * 32;
  const int wn = (wave >> 2) * 32;

  f32x8 acc[2][2] = {};

  // cooperative tile staging map
  const int arow = tid >> 1, ach = tid & 1;   // A tile: 128 rows x 2 chunks(16h)
  const int brow = tid >> 2, bch = tid & 3;   // B tile:  64 rows x 4 chunks(8h)
  const unsigned aLds = (unsigned)((arow * LDA_PAD + ach * 16) * 2);  // bytes
  const unsigned bLds = (unsigned)((brow * LDA_PAD + bch * 8) * 2);
  const unsigned asB[2] = { (unsigned)(uintptr_t)&As[0][0], (unsigned)(uintptr_t)&As[1][0] };
  const unsigned bsB[2] = { (unsigned)(uintptr_t)&Bs[0][0], (unsigned)(uintptr_t)&Bs[1][0] };
  const _Float16* aG = A + (size_t)(m0 + arow) * K + ach * 16;
  const _Float16* bG = W + (size_t)(n0 + brow) * K + bch * 8;

  auto issue = [&](int kk, int buf) {
    // 3 async b128 per wave per tile: 32B/thread of A + 16B/thread of W
    async_ld_b128(asB[buf] + aLds,      aG + kk);
    async_ld_b128(asB[buf] + aLds + 16, aG + kk + 8);
    async_ld_b128(bsB[buf] + bLds,      bG + kk);
  };

  const int T = K >> 5;
  issue(0, 0);
  for (int t = 0; t < T; ++t) {
    const int cur = t & 1;
    if (t + 1 < T) {
      issue((t + 1) << 5, cur ^ 1);
      // retire only the older tile's 3 asyncs; next tile stays in flight
      asm volatile("s_wait_asynccnt 0x3" ::: "memory");
    } else {
      asm volatile("s_wait_asynccnt 0x0" ::: "memory");
    }
    __syncthreads();   // every wave's tile-t asyncs retired -> tile visible

    const _Float16* as = As[cur];
    const _Float16* bs = Bs[cur];

    // --- build fragments per ISA 16-bit layouts ---
    f16x16 afrag[2], bfrag[2];
    #pragma unroll
    for (int i = 0; i < 2; i++) {
      const _Float16* p = as + (wm + i * 16 + l16) * LDA_PAD + lhalf * 8;
      union { f16x16 v; uint4 u[2]; } tu;
      tu.u[0] = *(const uint4*)(p);        // K = base .. base+7
      tu.u[1] = *(const uint4*)(p + 16);   // K = base+16 .. base+23
      afrag[i] = tu.v;
    }
    #pragma unroll
    for (int j = 0; j < 2; j++) {
      const _Float16* p = bs + (wn + j * 16 + l16) * LDA_PAD + lhalf * 8;
      union { f16x16 v; uint4 u[2]; } tu;
      tu.u[0] = *(const uint4*)(p);
      tu.u[1] = *(const uint4*)(p + 16);
      bfrag[j] = tu.v;
    }

    #pragma unroll
    for (int i = 0; i < 2; i++)
      #pragma unroll
      for (int j = 0; j < 2; j++)
        acc[i][j] = __builtin_amdgcn_wmma_f32_16x16x32_f16(
            false, afrag[i], false, bfrag[j], (short)0, acc[i][j], false, false);

    __syncthreads();   // all reads done before buffer is overwritten at t+1
  }

  // --- store C per 32-bit C/D layout: VGPR r -> row r (+8 for lanes 16-31) ---
  #pragma unroll
  for (int i = 0; i < 2; i++)
    #pragma unroll
    for (int j = 0; j < 2; j++) {
      const int col = n0 + wn + j * 16 + l16;
      float* cp = C + (size_t)(m0 + wm + i * 16 + lhalf * 8) * N + col;
      #pragma unroll
      for (int r = 0; r < 8; r++)
        cp[(size_t)r * N] = acc[i][j][r];
    }
}

// ---------------------------------------------------------------------------
// Elementwise / epilogue kernels
// ---------------------------------------------------------------------------
__global__ void k_cast(const float* __restrict__ in, _Float16* __restrict__ out, int n){
  int i = blockIdx.x * 256 + threadIdx.x;
  if (i < n) out[i] = (_Float16)in[i];
}

// Wb1/Wb2 masked cast: [8][256][256], mh[r][c] = (r%7+1 >= c%7+1)
__global__ void k_cast_mh(const float* __restrict__ in, _Float16* __restrict__ out, int n){
  int i = blockIdx.x * 256 + threadIdx.x;
  if (i >= n) return;
  int r = (i >> 8) & 255, c = i & 255;
  float m = ((r % 7) >= (c % 7)) ? 1.f : 0.f;
  out[i] = (_Float16)(in[i] * m);
}

// Wf masked + padded cast: in [L][184][256] -> out [L][192][256]
__global__ void k_cast_wf(const float* __restrict__ in, _Float16* __restrict__ out){
  int i = blockIdx.x * 256 + threadIdx.x;
  const int total = L_ * NPAD_ * H_;
  if (i >= total) return;
  int l = i / (NPAD_ * H_);
  int rem = i - l * NPAD_ * H_;
  int o = rem / H_, c = rem % H_;
  float v = 0.f;
  if (o < FD_ * MP_) {
    int od = o / MP_ + 1;          // out_deg
    int hd = c % 7 + 1;            // hid_deg
    if (od > hd) v = in[((size_t)l * FD_ * MP_ + o) * H_ + c];
  }
  out[i] = (_Float16)v;
}

__global__ void k_copy_f32(const float* __restrict__ in, float* __restrict__ out, int n){
  int i = blockIdx.x * 256 + threadIdx.x;
  if (i < n) out[i] = in[i];
}

// relu(layernorm(pre + bias)) -> f16, N fixed = 512 (2 elems / thread)
__global__ __launch_bounds__(256) void k_ln_relu_cast(
    const float* __restrict__ pre, const float* __restrict__ bias,
    const float* __restrict__ g, const float* __restrict__ bg,
    _Float16* __restrict__ out, int ldo, int outOff)
{
  __shared__ float s1[256], s2[256];
  const int t = threadIdx.x;
  const int row = blockIdx.x;
  const float* p = pre + (size_t)row * HID_;
  float v0 = p[t] + bias[t];
  float v1 = p[t + 256] + bias[t + 256];
  s1[t] = v0 + v1;
  s2[t] = v0 * v0 + v1 * v1;
  __syncthreads();
  for (int off = 128; off > 0; off >>= 1) {
    if (t < off) { s1[t] += s1[t + off]; s2[t] += s2[t + off]; }
    __syncthreads();
  }
  const float mu  = s1[0] * (1.0f / HID_);
  const float var = s2[0] * (1.0f / HID_) - mu * mu;
  const float rs  = rsqrtf(var + LNEPS);
  _Float16* o = out + (size_t)row * ldo + outOff;
  float r0 = (v0 - mu) * rs * g[t]       + bg[t];
  float r1 = (v1 - mu) * rs * g[t + 256] + bg[t + 256];
  o[t]       = (_Float16)fmaxf(r0, 0.f);
  o[t + 256] = (_Float16)fmaxf(r1, 0.f);
}

__global__ void k_ctx_fin(const float* __restrict__ pre, const float* __restrict__ bias,
                          float* __restrict__ ctx, _Float16* __restrict__ ctx16){
  int i = blockIdx.x * 256 + threadIdx.x;
  if (i >= B_ * CTX_) return;
  int nn = i % CTX_;
  float v = pre[i] + bias[nn];
  ctx[i] = v; ctx16[i] = (_Float16)v;
}

// h = ctx_part + z @ (W0*m0)^T + b0 + bc0 ; also write relu(h) f16
__global__ void k_h_init(float* __restrict__ h, const float* __restrict__ z,
    const float* __restrict__ W0, const float* __restrict__ b0,
    const float* __restrict__ bc0, _Float16* __restrict__ hrelu)
{
  int i = blockIdx.x * 256 + threadIdx.x;
  if (i >= B_ * H_) return;
  int b = i / H_, nn = i % H_;
  int deg = nn % 7 + 1;                 // hid_deg; m0[nn][k] = (k < deg)
  const float* zr = z + (size_t)b * FD_;
  const float* w  = W0 + (size_t)nn * FD_;
  float dot = 0.f;
  for (int k = 0; k < deg; k++) dot += zr[k] * w[k];
  float v = h[i] + dot + b0[nn] + bc0[nn];
  h[i] = v;
  hrelu[i] = (_Float16)fmaxf(v, 0.f);
}

__global__ void k_bias_relu_cast(const float* __restrict__ t, const float* __restrict__ bias,
                                 _Float16* __restrict__ out){
  int i = blockIdx.x * 256 + threadIdx.x;
  if (i >= B_ * H_) return;
  int nn = i % H_;
  out[i] = (_Float16)fmaxf(t[i] + bias[nn], 0.f);
}

// h += (t + bb2) * sigmoid(c + bcb); write relu(h) and raw h as f16
__global__ void k_combine(float* __restrict__ h, const float* __restrict__ t,
    const float* __restrict__ bb2, const float* __restrict__ c,
    const float* __restrict__ bcb, _Float16* __restrict__ hrelu,
    _Float16* __restrict__ hraw)
{
  int i = blockIdx.x * 256 + threadIdx.x;
  if (i >= B_ * H_) return;
  int nn = i % H_;
  float cc  = c[i] + bcb[nn];
  float sig = 1.0f / (1.0f + expf(-cc));
  float v = h[i] + (t[i] + bb2[nn]) * sig;
  h[i] = v;
  hrelu[i] = (_Float16)fmaxf(v, 0.f);
  hraw[i]  = (_Float16)v;
}

// LU inverse step: z = (z[perm] @ up^T) @ lw^T + b ; logq += sum(log(udiag))
__global__ __launch_bounds__(256) void k_lu(
    float* __restrict__ z, const float* __restrict__ Lm,
    const float* __restrict__ Um, const float* __restrict__ ld,
    const float* __restrict__ lb, const int* __restrict__ perm,
    float* __restrict__ logq, int first)
{
  int b = blockIdx.x * 256 + threadIdx.x;
  if (b >= B_) return;
  float zr[FD_], zp[FD_], y[FD_], udg[FD_];
  #pragma unroll
  for (int k = 0; k < FD_; k++) zr[k] = z[(size_t)b * FD_ + k];
  #pragma unroll
  for (int k = 0; k < FD_; k++) zp[k] = zr[perm[k]];
  float ldet = 0.f;
  #pragma unroll
  for (int j = 0; j < FD_; j++) { udg[j] = softplusf(ld[j]) + 0.001f; ldet += logf(udg[j]); }
  #pragma unroll
  for (int j = 0; j < FD_; j++) {
    float a = zp[j] * udg[j];
    for (int k = j + 1; k < FD_; k++) a += Um[j * FD_ + k] * zp[k];
    y[j] = a;
  }
  #pragma unroll
  for (int j = 0; j < FD_; j++) {
    float a = y[j];
    for (int k = 0; k < j; k++) a += Lm[j * FD_ + k] * y[k];
    z[(size_t)b * FD_ + j] = a + lb[j];
  }
  if (first) logq[b] = ldet; else logq[b] += ldet;
}

// rational-quadratic spline forward, per row; updates z, accumulates logq
__global__ __launch_bounds__(256) void k_spline(
    float* __restrict__ z, const float* __restrict__ params,
    const float* __restrict__ bfp, float* __restrict__ logq)
{
  int b = blockIdx.x * 256 + threadIdx.x;
  if (b >= B_) return;
  const float* pr = params + (size_t)b * NPAD_;
  float zr[FD_];
  #pragma unroll
  for (int f = 0; f < FD_; f++) zr[f] = z[(size_t)b * FD_ + f];
  float ladsum = 0.f;
  for (int f = 0; f < FD_; f++) {
    const int base = f * MP_;
    float uw[8], uh[8], cw[9], chh[9], wv[8], hv[8], dd[9];
    float mw = -1e30f, mh = -1e30f;
    for (int k = 0; k < 8; k++) {
      uw[k] = (pr[base + k]     + bfp[base + k])     * (1.0f / 16.0f);  // / sqrt(H)
      uh[k] = (pr[base + 8 + k] + bfp[base + 8 + k]) * (1.0f / 16.0f);
      mw = fmaxf(mw, uw[k]); mh = fmaxf(mh, uh[k]);
    }
    float sw = 0.f, sh = 0.f;
    for (int k = 0; k < 8; k++) {
      uw[k] = expf(uw[k] - mw); sw += uw[k];
      uh[k] = expf(uh[k] - mh); sh += uh[k];
    }
    float accw = 0.f, acch = 0.f;
    cw[0] = -TBND; chh[0] = -TBND;
    for (int k = 0; k < 8; k++) {
      float wk = 0.001f + (1.0f - 0.008f) * (uw[k] / sw);
      float hk = 0.001f + (1.0f - 0.008f) * (uh[k] / sh);
      accw += wk; acch += hk;
      cw[k + 1]  = 2.0f * TBND * accw - TBND;
      chh[k + 1] = 2.0f * TBND * acch - TBND;
    }
    cw[8] = TBND; chh[8] = TBND;
    for (int k = 0; k < 8; k++) { wv[k] = cw[k + 1] - cw[k]; hv[k] = chh[k + 1] - chh[k]; }
    dd[0] = 1.0f; dd[8] = 1.0f;   // MD + softplus(log(expm1(1-MD))) == 1
    for (int k = 0; k < 7; k++)
      dd[k + 1] = 0.001f + softplusf(pr[base + 16 + k] + bfp[base + 16 + k]);

    float x = zr[f];
    bool inside = (x >= -TBND) && (x <= TBND);
    float xc = fminf(fmaxf(x, -TBND), TBND);
    int cnt = 0;
    for (int k = 0; k < 9; k++) cnt += (xc >= cw[k]) ? 1 : 0;
    int idx = cnt - 1; idx = idx < 0 ? 0 : (idx > 7 ? 7 : idx);
    float icw = cw[idx], iw = wv[idx], ich = chh[idx], ih = hv[idx];
    float dk = dd[idx], dk1 = dd[idx + 1];
    float delta = ih / iw;
    float th  = (xc - icw) / iw;
    float th1 = th * (1.0f - th);
    float den = delta + (dk + dk1 - 2.0f * delta) * th1;
    float out = ich + ih * (delta * th * th + dk * th1) / den;
    float omt = 1.0f - th;
    float dnum = delta * delta * (dk1 * th * th + 2.0f * delta * th1 + dk * omt * omt);
    float lad = logf(dnum) - 2.0f * logf(den);
    zr[f] = inside ? out : x;
    ladsum += inside ? lad : 0.0f;
  }
  #pragma unroll
  for (int f = 0; f < FD_; f++) z[(size_t)b * FD_ + f] = zr[f];
  logq[b] += ladsum;
}

__global__ __launch_bounds__(256) void k_rowfinal(
    const float* __restrict__ z, const float* __restrict__ logq,
    float* __restrict__ partials)
{
  __shared__ float s[256];
  int b = blockIdx.x * 256 + threadIdx.x;
  float ss = 0.f;
  #pragma unroll
  for (int k = 0; k < FD_; k++) { float v = z[(size_t)b * FD_ + k]; ss += v * v; }
  s[threadIdx.x] = logq[b] - 7.3515083f - 0.5f * ss;   // 0.5*FD*log(2*pi)
  __syncthreads();
  for (int off = 128; off > 0; off >>= 1) {
    if (threadIdx.x < off) s[threadIdx.x] += s[threadIdx.x + off];
    __syncthreads();
  }
  if (threadIdx.x == 0) partials[blockIdx.x] = s[0];
}

__global__ void k_final(const float* __restrict__ partials, float* __restrict__ out){
  __shared__ float s[64];
  int t = threadIdx.x;
  s[t] = partials[t];
  __syncthreads();
  for (int off = 32; off > 0; off >>= 1) {
    if (t < off) s[t] += s[t + off];
    __syncthreads();
  }
  if (t == 0) out[0] = -(s[0] / (float)B_);
}

// ---------------------------------------------------------------------------
// Host orchestration
// ---------------------------------------------------------------------------
static void gemm(const _Float16* A, const _Float16* W, float* C,
                 int M, int N, int K, hipStream_t s){
  dim3 grid(N / 64, M / 128);
  k_gemm<<<grid, 256, 0, s>>>(A, W, C, M, N, K);
}

extern "C" void kernel_launch(void* const* d_in, const int* in_sizes, int n_in,
                              void* d_out, int out_size, void* d_ws, size_t ws_size,
                              hipStream_t stream)
{
  (void)in_sizes; (void)n_in; (void)out_size; (void)ws_size;
  // ---- inputs (leaf-flattened in setup_inputs dict order) ----
  const float* state  = (const float*)d_in[0];
  const float* action = (const float*)d_in[1];
  const float* x_pad  = (const float*)d_in[2];
  const float* Ws  = (const float*)d_in[3];
  const float* bs  = (const float*)d_in[4];
  const float* gs  = (const float*)d_in[5];
  const float* bgs = (const float*)d_in[6];
  const float* Wa  = (const float*)d_in[7];
  const float* ba  = (const float*)d_in[8];
  const float* ga  = (const float*)d_in[9];
  const float* bga = (const float*)d_in[10];
  const float* Wf1 = (const float*)d_in[11];
  const float* bf1 = (const float*)d_in[12];
  const float* gf1 = (const float*)d_in[13];
  const float* bgf1= (const float*)d_in[14];
  const float* Wf2 = (const float*)d_in[15];
  const float* bf2 = (const float*)d_in[16];
  const float* gf2 = (const float*)d_in[17];
  const float* bgf2= (const float*)d_in[18];
  const float* Wo1 = (const float*)d_in[19];
  const float* bo1 = (const float*)d_in[20];
  const float* go1 = (const float*)d_in[21];
  const float* bgo1= (const float*)d_in[22];
  const float* Wo2 = (const float*)d_in[23];
  const float* bo2 = (const float*)d_in[24];
  const float* W0  = (const float*)d_in[25];
  const float* b0  = (const float*)d_in[26];
  const float* Wc0 = (const float*)d_in[27];
  const float* bc0 = (const float*)d_in[28];
  const float* Wb1 = (const float*)d_in[29];
  const float* bb1 = (const float*)d_in[30];
  const float* Wb2 = (const float*)d_in[31];
  const float* bb2 = (const float*)d_in[32];
  const float* Wcb = (const float*)d_in[33];
  const float* bcb = (const float*)d_in[34];
  const float* Wff = (const float*)d_in[35];
  const float* bff = (const float*)d_in[36];
  const float* lu_L = (const float*)d_in[37];
  const float* lu_U = (const float*)d_in[38];
  const float* lu_d = (const float*)d_in[39];
  const float* lu_b = (const float*)d_in[40];
  const int*   perms = (const int*)d_in[41];

  // ---- workspace layout ----
  char* base = (char*)d_ws;
  size_t off = 0;
  auto alloc = [&](size_t bytes) -> char* {
    char* p = base + off;
    off += (bytes + 255) & ~(size_t)255;
    return p;
  };
  _Float16* Wsf16  = (_Float16*)alloc((size_t)HID_*SD_*2);
  _Float16* Waf16  = (_Float16*)alloc((size_t)HID_*AD_*2);
  _Float16* Wf1f16 = (_Float16*)alloc((size_t)HID_*2*HID_*2);
  _Float16* Wf2f16 = (_Float16*)alloc((size_t)HID_*HID_*2);
  _Float16* Wo1f16 = (_Float16*)alloc((size_t)HID_*HID_*2);
  _Float16* Wo2f16 = (_Float16*)alloc((size_t)HID_*HID_*2);
  _Float16* Wc0f16 = (_Float16*)alloc((size_t)L_*H_*CTX_*2);
  _Float16* Wb1f16 = (_Float16*)alloc((size_t)L_*NB_*H_*H_*2);
  _Float16* Wb2f16 = (_Float16*)alloc((size_t)L_*NB_*H_*H_*2);
  _Float16* Wcbf16 = (_Float16*)alloc((size_t)L_*NB_*H_*CTX_*2);
  _Float16* Wff16  = (_Float16*)alloc((size_t)L_*NPAD_*H_*2);
  _Float16* statef16  = (_Float16*)alloc((size_t)B_*SD_*2);
  _Float16* actionf16 = (_Float16*)alloc((size_t)B_*AD_*2);
  _Float16* h0f16  = (_Float16*)alloc((size_t)B_*2*HID_*2);
  _Float16* actf16 = (_Float16*)alloc((size_t)B_*HID_*2);
  _Float16* ctxf16 = (_Float16*)alloc((size_t)B_*CTX_*2);
  _Float16* hRelu16= (_Float16*)alloc((size_t)B_*H_*2);
  _Float16* hRaw16 = (_Float16*)alloc((size_t)B_*H_*2);
  _Float16* t16    = (_Float16*)alloc((size_t)B_*H_*2);
  float* preF32    = (float*)alloc((size_t)B_*HID_*4);
  float* ctxF32    = (float*)alloc((size_t)B_*CTX_*4);
  float* hF32      = (float*)alloc((size_t)B_*H_*4);
  float* tF32      = (float*)alloc((size_t)B_*H_*4);
  float* cF32      = (float*)alloc((size_t)B_*H_*4);
  float* paramsF32 = (float*)alloc((size_t)B_*NPAD_*4);
  float* zbuf      = (float*)alloc((size_t)B_*FD_*4);
  float* logq      = (float*)alloc((size_t)B_*4);
  float* partials  = (float*)alloc(64*4);

  auto cast = [&](const float* in, _Float16* out, int n){
    k_cast<<<(n + 255) / 256, 256, 0, stream>>>(in, out, n);
  };

  // ---- weight casts (masked where needed) ----
  cast(Ws,  Wsf16,  HID_*SD_);
  cast(Wa,  Waf16,  HID_*AD_);
  cast(Wf1, Wf1f16, HID_*2*HID_);
  cast(Wf2, Wf2f16, HID_*HID_);
  cast(Wo1, Wo1f16, HID_*HID_);
  cast(Wo2, Wo2f16, HID_*HID_);
  cast(Wc0, Wc0f16, L_*H_*CTX_);
  cast(Wcb, Wcbf16, L_*NB_*H_*CTX_);
  {
    int n = L_*NB_*H_*H_;
    k_cast_mh<<<(n + 255) / 256, 256, 0, stream>>>(Wb1, Wb1f16, n);
    k_cast_mh<<<(n + 255) / 256, 256, 0, stream>>>(Wb2, Wb2f16, n);
    int nf = L_*NPAD_*H_;
    k_cast_wf<<<(nf + 255) / 256, 256, 0, stream>>>(Wff, Wff16);
  }
  cast(state,  statef16,  B_*SD_);
  cast(action, actionf16, B_*AD_);

  // ---- encoder ----
  gemm(statef16, Wsf16, preF32, B_, HID_, SD_, stream);
  k_ln_relu_cast<<<B_, 256, 0, stream>>>(preF32, bs, gs, bgs, h0f16, 2*HID_, 0);
  gemm(actionf16, Waf16, preF32, B_, HID_, AD_, stream);
  k_ln_relu_cast<<<B_, 256, 0, stream>>>(preF32, ba, ga, bga, h0f16, 2*HID_, HID_);
  gemm(h0f16, Wf1f16, preF32, B_, HID_, 2*HID_, stream);
  k_ln_relu_cast<<<B_, 256, 0, stream>>>(preF32, bf1, gf1, bgf1, actf16, HID_, 0);
  gemm(actf16, Wf2f16, preF32, B_, HID_, HID_, stream);
  k_ln_relu_cast<<<B_, 256, 0, stream>>>(preF32, bf2, gf2, bgf2, actf16, HID_, 0);
  gemm(actf16, Wo1f16, preF32, B_, HID_, HID_, stream);
  k_ln_relu_cast<<<B_, 256, 0, stream>>>(preF32, bo1, go1, bgo1, actf16, HID_, 0);
  gemm(actf16, Wo2f16, preF32, B_, HID_, HID_, stream);
  k_ctx_fin<<<(B_*CTX_ + 255)/256, 256, 0, stream>>>(preF32, bo2, ctxF32, ctxf16);

  // ---- flow: z <- x_pad ----
  k_copy_f32<<<(B_*FD_ + 255)/256, 256, 0, stream>>>(x_pad, zbuf, B_*FD_);

  for (int i = L_ - 1; i >= 0; --i) {
    k_lu<<<B_/256, 256, 0, stream>>>(zbuf,
        lu_L + (size_t)i*FD_*FD_, lu_U + (size_t)i*FD_*FD_,
        lu_d + (size_t)i*FD_, lu_b + (size_t)i*FD_,
        perms + (size_t)i*FD_, logq, (i == L_ - 1) ? 1 : 0);

    gemm(ctxf16, Wc0f16 + (size_t)i*H_*CTX_, hF32, B_, H_, CTX_, stream);
    k_h_init<<<(B_*H_)/256, 256, 0, stream>>>(hF32, zbuf,
        W0 + (size_t)i*H_*FD_, b0 + (size_t)i*H_, bc0 + (size_t)i*H_, hRelu16);

    for (int j = 0; j < NB_; j++) {
      const size_t bi = (size_t)i*NB_ + j;
      gemm(hRelu16, Wb1f16 + bi*H_*H_, tF32, B_, H_, H_, stream);
      k_bias_relu_cast<<<(B_*H_)/256, 256, 0, stream>>>(tF32, bb1 + bi*H_, t16);
      gemm(t16, Wb2f16 + bi*H_*H_, tF32, B_, H_, H_, stream);
      gemm(ctxf16, Wcbf16 + bi*H_*CTX_, cF32, B_, H_, CTX_, stream);
      k_combine<<<(B_*H_)/256, 256, 0, stream>>>(hF32, tF32, bb2 + bi*H_,
          cF32, bcb + bi*H_, hRelu16, hRaw16);
    }

    gemm(hRaw16, Wff16 + (size_t)i*NPAD_*H_, paramsF32, B_, NPAD_, H_, stream);
    k_spline<<<B_/256, 256, 0, stream>>>(zbuf, paramsF32,
        bff + (size_t)i*FD_*MP_, logq);
  }

  k_rowfinal<<<B_/256, 256, 0, stream>>>(zbuf, logq, partials);
  k_final<<<1, 64, 0, stream>>>(partials, (float*)d_out);
}
